// MultiHeadSelfAttention_4939212390705
// MI455X (gfx1250) — compile-verified, matrix operand order checked by reading, not run on hardware
//
#include <hip/hip_runtime.h>
#include <hip/hip_bf16.h>

typedef __bf16 bf16_t;
typedef __attribute__((ext_vector_type(16))) bf16_t v16bf;
typedef __attribute__((ext_vector_type(8)))  bf16_t v8bf;
typedef __attribute__((ext_vector_type(8)))  float  v8f;

// ---------------------------------------------------------------------------
__device__ __forceinline__ v8f wmma_bf16(v16bf a, v16bf b, v8f c) {
  return __builtin_amdgcn_wmma_f32_16x16x32_bf16(
      /*neg_a=*/false, a, /*neg_b=*/false, b,
      /*c_mod=*/(short)0, c, /*reuse_a=*/false, /*reuse_b=*/false);
}

// A-style 16x32 fragment from row-major fp32 source (convert to bf16).
// Lane layout (ISA 7.12.2): row = lane&15; K runs [hi*8,+8) and [16+hi*8,+8).
__device__ __forceinline__ v16bf frag_ld_f32(const float* p0, int ld) {
  const int lane = threadIdx.x & 31;
  const int row = lane & 15, hi = lane >> 4;
  const float* p = p0 + (size_t)row * ld + hi * 8;
  float4 x0 = *(const float4*)(p);
  float4 x1 = *(const float4*)(p + 4);
  float4 y0 = *(const float4*)(p + 16);
  float4 y1 = *(const float4*)(p + 20);
  v16bf f;
  f[0]  = (bf16_t)x0.x; f[1]  = (bf16_t)x0.y; f[2]  = (bf16_t)x0.z; f[3]  = (bf16_t)x0.w;
  f[4]  = (bf16_t)x1.x; f[5]  = (bf16_t)x1.y; f[6]  = (bf16_t)x1.z; f[7]  = (bf16_t)x1.w;
  f[8]  = (bf16_t)y0.x; f[9]  = (bf16_t)y0.y; f[10] = (bf16_t)y0.z; f[11] = (bf16_t)y0.w;
  f[12] = (bf16_t)y1.x; f[13] = (bf16_t)y1.y; f[14] = (bf16_t)y1.z; f[15] = (bf16_t)y1.w;
  return f;
}

// A-style 16x32 fragment from row-major bf16 source (global or LDS).
__device__ __forceinline__ v16bf frag_ld_bf16(const bf16_t* p0, int ld) {
  const int lane = threadIdx.x & 31;
  const int row = lane & 15, hi = lane >> 4;
  const bf16_t* p = p0 + (size_t)row * ld + hi * 8;
  v8bf lo = *(const v8bf*)(p);
  v8bf hh = *(const v8bf*)(p + 16);
  return __builtin_shufflevector(lo, hh, 0,1,2,3,4,5,6,7,8,9,10,11,12,13,14,15);
}

// gfx1250 async copy: 16B global -> LDS, tracked by ASYNCcnt.
__device__ __forceinline__ void async_g2l_b128(unsigned lds_byte_addr, const void* g) {
  asm volatile("global_load_async_to_lds_b128 %0, %1, off"
               :: "v"(lds_byte_addr), "v"(g) : "memory");
}

// ---------------------------------------------------------------------------
// Kernel 1: qkv = x @ Wqkv^T + b ; scatter to Q (scaled), K, V^T as bf16.
// M=4096, N=3072, K=1024. Block = 64 rows x 256 cols; 8 waves of 32x64.
__global__ __launch_bounds__(256) void k_qkv(
    const float* __restrict__ x, const float* __restrict__ Wqkv,
    const float* __restrict__ bqkv,
    bf16_t* __restrict__ Qb, bf16_t* __restrict__ Kb, bf16_t* __restrict__ Vt) {
  const int w = threadIdx.x >> 5;
  const int lane = threadIdx.x & 31;
  const int hi = lane >> 4;
  const int rbase = blockIdx.x * 64 + (w & 1) * 32;
  const int cbase = blockIdx.y * 256 + (w >> 1) * 64;
  v8f acc[8] = {{}, {}, {}, {}, {}, {}, {}, {}};  // [row frag 0..1][col frag 0..3]
  for (int kk = 0; kk < 1024; kk += 32) {
    if (kk + 32 < 1024) {  // prefetch next K-slice of the weights (L2 hint)
      __builtin_prefetch(Wqkv + (size_t)cbase * 1024 + kk + 32, 0, 1);
      __builtin_prefetch(x + (size_t)rbase * 1024 + kk + 32, 0, 1);
    }
    v16bf a0 = frag_ld_f32(x + (size_t)rbase * 1024 + kk, 1024);
    v16bf a1 = frag_ld_f32(x + (size_t)(rbase + 16) * 1024 + kk, 1024);
#pragma unroll
    for (int j = 0; j < 4; j++) {
      v16bf bj = frag_ld_f32(Wqkv + (size_t)(cbase + 16 * j) * 1024 + kk, 1024);
      acc[j]     = wmma_bf16(a0, bj, acc[j]);
      acc[4 + j] = wmma_bf16(a1, bj, acc[4 + j]);
    }
  }
  // softmax scale (1/sqrt(64)) and log2(e) folded into Q so attention uses exp2
  const float QS = 0.125f * 1.44269504088896f;
#pragma unroll
  for (int q = 0; q < 8; q++) {
    int r0 = rbase + (q >> 2) * 16;
    int c0 = cbase + (q & 3) * 16;
    int col = c0 + (lane & 15);
    float bv = bqkv[col];
    int h = col / 192, t = col - h * 192;     // per-head interleaved q|k|v
    int which = t >> 6, d = t & 63;
#pragma unroll
    for (int e = 0; e < 8; e++) {
      int row = r0 + e + hi * 8;              // C layout: M = e (+8 for hi lanes)
      int b = row >> 11, s = row & 2047;
      size_t bh = (size_t)(b * 16 + h);
      float v = acc[q][e] + bv;
      if (which == 0)      Qb[(bh * 2048 + s) * 64 + d] = (bf16_t)(v * QS);
      else if (which == 1) Kb[(bh * 2048 + s) * 64 + d] = (bf16_t)v;
      else                 Vt[(bh * 64 + d) * 2048 + s] = (bf16_t)v;  // transposed
    }
  }
}

// ---------------------------------------------------------------------------
// Kernel 2: flash attention. 1 block = one (bh, 64-query tile); 4 waves x 16 rows.
// K/V tiles staged to LDS via async copies, double-buffered, shared by all waves.
__global__ __launch_bounds__(128) void k_attn(
    const bf16_t* __restrict__ Qb, const bf16_t* __restrict__ Kb,
    const bf16_t* __restrict__ Vt, bf16_t* __restrict__ attn) {
  __shared__ __align__(16) bf16_t Kl[2][32][64];   // keys x features (8KB)
  __shared__ __align__(16) bf16_t Vl[2][64][32];   // features x keys (8KB)
  __shared__ __align__(16) bf16_t Plds[4][16][32]; // per-wave P transpose (4KB)
  const int tid = threadIdx.x;
  const int w = tid >> 5;
  const int lane = tid & 31;
  const int hi = lane >> 4;
  const int bh = blockIdx.x >> 5;
  const int qt = blockIdx.x & 31;
  const bf16_t* Q = Qb + (size_t)bh * 2048 * 64;
  const bf16_t* K = Kb + (size_t)bh * 2048 * 64;
  const bf16_t* V = Vt + (size_t)bh * 64 * 2048;
  const int qrow0 = qt * 64 + w * 16;

  // stage K[kt..kt+31][0..63] and V^T[0..63][kt..kt+31] into buffer bufi
  auto stage = [&](int kt, int bufi) {
    // K tile: 32 rows * 128B = 256 x 16B chunks (8 per row)
    for (int c = tid; c < 256; c += 128) {
      unsigned lds = (unsigned)(uintptr_t)&Kl[bufi][0][0] + c * 16;
      const char* g = (const char*)(K + (size_t)(kt + (c >> 3)) * 64) + (c & 7) * 16;
      async_g2l_b128(lds, g);
    }
    // V tile: 64 rows * 64B = 256 x 16B chunks (4 per row)
    for (int c = tid; c < 256; c += 128) {
      unsigned lds = (unsigned)(uintptr_t)&Vl[bufi][0][0] + c * 16;
      const char* g = (const char*)(V + (size_t)(c >> 2) * 2048 + kt) + (c & 3) * 16;
      async_g2l_b128(lds, g);
    }
  };

  v16bf qa0 = frag_ld_bf16(Q + (size_t)qrow0 * 64, 64);       // K-dim 0..31
  v16bf qa1 = frag_ld_bf16(Q + (size_t)qrow0 * 64 + 32, 64);  // K-dim 32..63
  v8f acc[4] = {{}, {}, {}, {}};
  float m[8], l[8];
#pragma unroll
  for (int e = 0; e < 8; e++) { m[e] = -1e30f; l[e] = 0.0f; }

  stage(0, 0);
  for (int kt = 0; kt < 2048; kt += 32) {
    const int p = (kt >> 5) & 1;
    if (kt + 32 < 2048) {
      stage(kt + 32, p ^ 1);                         // 4 async instr per wave
      asm volatile("s_wait_asynccnt 4" ::: "memory");// batch for buf[p] landed
    } else {
      asm volatile("s_wait_asynccnt 0" ::: "memory");
    }
    __syncthreads();                                  // visible to all waves
    const bf16_t* Kt = &Kl[p][0][0];
    const bf16_t* Vb = &Vl[p][0][0];

    // scores for 32 keys: two 16-key accumulators, K-dim 64 in two steps
    v16bf k00 = frag_ld_bf16(Kt, 64);
    v16bf k01 = frag_ld_bf16(Kt + 32, 64);
    v16bf k10 = frag_ld_bf16(Kt + 16 * 64, 64);
    v16bf k11 = frag_ld_bf16(Kt + 16 * 64 + 32, 64);
    v8f s0 = {}, s1 = {};
    s0 = wmma_bf16(qa0, k00, s0); s0 = wmma_bf16(qa1, k01, s0);
    s1 = wmma_bf16(qa0, k10, s1); s1 = wmma_bf16(qa1, k11, s1);

    // online softmax; C-layout rows live in 16-lane groups -> xor 1,2,4,8
    float p0[8], p1[8], al[8];
#pragma unroll
    for (int e = 0; e < 8; e++) {
      float t = fmaxf(s0[e], s1[e]);
#pragma unroll
      for (int msk = 1; msk < 16; msk <<= 1) t = fmaxf(t, __shfl_xor(t, msk, 32));
      float mn = fmaxf(m[e], t);
      al[e] = exp2f(m[e] - mn);
      m[e] = mn;
      p0[e] = exp2f(s0[e] - mn);
      p1[e] = exp2f(s1[e] - mn);
      float rs = p0[e] + p1[e];
#pragma unroll
      for (int msk = 1; msk < 16; msk <<= 1) rs += __shfl_xor(rs, msk, 32);
      l[e] = l[e] * al[e] + rs;
    }
#pragma unroll
    for (int c = 0; c < 4; c++)
#pragma unroll
      for (int e = 0; e < 8; e++) acc[c][e] *= al[e];

    // C-layout P -> LDS -> A-layout fragment (16x32)
#pragma unroll
    for (int e = 0; e < 8; e++) {
      int rl = e + hi * 8;
      Plds[w][rl][lane & 15] = (bf16_t)p0[e];
      Plds[w][rl][16 + (lane & 15)] = (bf16_t)p1[e];
    }
    asm volatile("s_wait_dscnt 0" ::: "memory");   // cross-lane LDS RAW, same wave
    v16bf pa = frag_ld_bf16(&Plds[w][0][0], 32);
#pragma unroll
    for (int c = 0; c < 4; c++) {
      v16bf vf = frag_ld_bf16(Vb + (size_t)(c * 16) * 32, 32);
      acc[c] = wmma_bf16(pa, vf, acc[c]);
    }
    __syncthreads();   // all waves done with buf[p] before it is overwritten
  }

  // normalize and store to attn [B*S, 1024] bf16
  const int b = bh >> 4, h = bh & 15;
#pragma unroll
  for (int e = 0; e < 8; e++) {
    float inv = 1.0f / l[e];
    int srow = qrow0 + e + hi * 8;
    size_t rb = ((size_t)b * 2048 + srow) * 1024;
#pragma unroll
    for (int c = 0; c < 4; c++) {
      int col = h * 64 + c * 16 + (lane & 15);
      attn[rb + col] = (bf16_t)(acc[c][e] * inv);
    }
  }
}

// ---------------------------------------------------------------------------
// Kernel 3: out = attn @ Wout^T + b_out. M=4096, N=1024, K=1024, fp32 output.
// Block = 64 x 256; 8 waves of 32x64.
__global__ __launch_bounds__(256) void k_out(
    const bf16_t* __restrict__ A, const float* __restrict__ W,
    const float* __restrict__ bias, float* __restrict__ out) {
  const int w = threadIdx.x >> 5;
  const int lane = threadIdx.x & 31;
  const int hi = lane >> 4;
  const int rbase = blockIdx.x * 64 + (w & 1) * 32;
  const int cbase = blockIdx.y * 256 + (w >> 1) * 64;
  v8f acc[8] = {{}, {}, {}, {}, {}, {}, {}, {}};
  for (int kk = 0; kk < 1024; kk += 32) {
    if (kk + 32 < 1024) {
      __builtin_prefetch(W + (size_t)cbase * 1024 + kk + 32, 0, 1);
    }
    v16bf a0 = frag_ld_bf16(A + (size_t)rbase * 1024 + kk, 1024);
    v16bf a1 = frag_ld_bf16(A + (size_t)(rbase + 16) * 1024 + kk, 1024);
#pragma unroll
    for (int j = 0; j < 4; j++) {
      v16bf bj = frag_ld_f32(W + (size_t)(cbase + 16 * j) * 1024 + kk, 1024);
      acc[j]     = wmma_bf16(a0, bj, acc[j]);
      acc[4 + j] = wmma_bf16(a1, bj, acc[4 + j]);
    }
  }
#pragma unroll
  for (int q = 0; q < 8; q++) {
    int r0 = rbase + (q >> 2) * 16;
    int c0 = cbase + (q & 3) * 16;
    int col = c0 + (lane & 15);
    float bv = bias[col];
#pragma unroll
    for (int e = 0; e < 8; e++) {
      int row = r0 + e + hi * 8;
      out[(size_t)row * 1024 + col] = acc[q][e] + bv;
    }
  }
}

// ---------------------------------------------------------------------------
extern "C" void kernel_launch(void* const* d_in, const int* in_sizes, int n_in,
                              void* d_out, int out_size, void* d_ws, size_t ws_size,
                              hipStream_t stream) {
  (void)in_sizes; (void)n_in; (void)out_size; (void)ws_size;
  const float* x    = (const float*)d_in[0];
  const float* Wqkv = (const float*)d_in[1];
  const float* bqkv = (const float*)d_in[2];
  const float* Wout = (const float*)d_in[3];
  const float* bout = (const float*)d_in[4];
  float* out = (float*)d_out;

  // bf16 workspace: Q [32,2048,64] | K [32,2048,64] | V^T [32,64,2048] | attn [4096,1024]
  bf16_t* Qb = (bf16_t*)d_ws;
  bf16_t* Kb = Qb + (size_t)32 * 2048 * 64;
  bf16_t* Vt = Kb + (size_t)32 * 2048 * 64;
  bf16_t* At = Vt + (size_t)32 * 2048 * 64;

  k_qkv<<<dim3(4096 / 64, 3072 / 256), 256, 0, stream>>>(x, Wqkv, bqkv, Qb, Kb, Vt);
  k_attn<<<dim3(32 * 32), 128, 0, stream>>>(Qb, Kb, Vt, At);
  k_out<<<dim3(4096 / 64, 1024 / 256), 256, 0, stream>>>(At, Wout, bout, out);
}
